// RNNNet_67095979099001
// MI455X (gfx1250) — compile-verified, matrix operand order checked by reading, not run on hardware
//
#include <hip/hip_runtime.h>

typedef __attribute__((ext_vector_type(16))) _Float16 v16h;
typedef __attribute__((ext_vector_type(8)))  _Float16 v8h;
typedef __attribute__((ext_vector_type(8)))  float    v8f;
typedef __attribute__((ext_vector_type(4)))  float    v4f;

#define B_ 64
#define T_ 16384
#define I_ 28
#define H_ 64
#define C_ 10

// Workspace layout (floats):
//   [0, T_*4096)             xp tiles, layout [t][mblk(4)][nblk(4)][lane(32)][v(8)]
//                            (exactly the WMMA f32 16x16 C-fragment layout)
//   [T_*4096, T_*4096+4096)  final hidden state h_T, row-major [64][64]

// ---------------------------------------------------------------------------
// Kernel 1: x_proj[b,t,h] = input[b,t,:] . W_ih[h,:] + b_ih[h] + b_hh[h],
// stored pre-permuted into C-fragment layout per timestep tile.
// ---------------------------------------------------------------------------
__global__ __launch_bounds__(256) void xproj_kernel(
    const float* __restrict__ input, const float* __restrict__ W_ih,
    const float* __restrict__ b_ih, const float* __restrict__ b_hh,
    float* __restrict__ ws) {
  __shared__ float in_sh[B_ * I_];   // 1792 floats
  __shared__ float w_sh[H_ * I_];    // 1792 floats
  __shared__ float bias_sh[H_];
  const int t = blockIdx.x;
  const int tid = threadIdx.x;

  for (int idx = tid; idx < B_ * I_; idx += 256) {
    int b = idx / I_, i = idx - b * I_;
    in_sh[idx] = input[((size_t)b * T_ + t) * I_ + i];
  }
  for (int idx = tid; idx < H_ * I_; idx += 256) w_sh[idx] = W_ih[idx];
  if (tid < H_) bias_sh[tid] = b_ih[tid] + b_hh[tid];
  __syncthreads();

  float* out = ws + (size_t)t * 4096;
  for (int e = tid; e < 4096; e += 256) {
    int m = e >> 10, n = (e >> 8) & 3, lane = (e >> 3) & 31, v = e & 7;
    int b = m * 16 + v + ((lane >> 4) << 3);  // C layout: M = v + 8*(lane>=16)
    int h = n * 16 + (lane & 15);             // C layout: N = lane & 15
    float s = bias_sh[h];
    const float* ir = in_sh + b * I_;
    const float* wr = w_sh + h * I_;
#pragma unroll
    for (int i = 0; i < I_; ++i) s = fmaf(ir[i], wr[i], s);
    out[e] = s;
  }
}

// ---------------------------------------------------------------------------
// Kernel 2: sequential scan. 4 workgroups (one per 16-row batch block),
// 4 waves each; wave w owns output columns [16w, 16w+16).
// ---------------------------------------------------------------------------
__device__ __forceinline__ float fast_tanh(float x) {
  float xc = fminf(fmaxf(x, -9.0f), 9.0f);
  float e = __builtin_amdgcn_exp2f(xc * 2.8853900817779268f);  // e^(2x)
  return (e - 1.0f) * __builtin_amdgcn_rcpf(e + 1.0f);
}

__global__ __launch_bounds__(128) void rnn_scan_kernel(
    const float* __restrict__ W_hh, const float* __restrict__ ws_xp,
    float* __restrict__ hfinal) {
  __shared__ __align__(16) _Float16 hbuf[2][16 * 64];  // double-buffered h (f16)

  const int lane = threadIdx.x & 31;
  const int nblk = threadIdx.x >> 5;  // 0..3 : output column tile
  const int mblk = blockIdx.x;        // 0..3 : batch row block
  const int nloc = lane & 15;
  const int koff = (lane >> 4) << 3;  // 0 or 8 (fragment K sub-offset)

  // h0 = 0
  for (int idx = threadIdx.x; idx < 16 * 64; idx += 128)
    hbuf[0][idx] = (_Float16)0.0f;

  // Resident B fragments: B[k][n] = W_hh[n_global][k] (row of W_hh is
  // contiguous in k, so each lane reads two 8-float chunks per fragment).
  const int nglob = nblk * 16 + nloc;
  const float* wrow = W_hh + nglob * H_;
  v16h Bf[2];
#pragma unroll
  for (int f = 0; f < 2; ++f) {
    const int kb = f * 32;
#pragma unroll
    for (int j = 0; j < 8; ++j) {
      Bf[f][j]     = (_Float16)wrow[kb + koff + j];
      Bf[f][8 + j] = (_Float16)wrow[kb + 16 + koff + j];
    }
  }

  const int r = nloc;  // A-fragment row (lanes 0-15 and 16-31 both map M=0..15)
  const float* xpbase =
      ws_xp + (size_t)(mblk * 4 + nblk) * 256 + (size_t)lane * 8;

  union PUN8 { v8f v; v4f q[2]; };

  // 8-deep register software pipeline on the xp stream.
  constexpr int PF = 8;
  v8f pf[PF];
#pragma unroll
  for (int p = 0; p < PF; ++p) {
    const v4f* src = (const v4f*)(xpbase + (size_t)p * 4096);
    PUN8 u; u.q[0] = src[0]; u.q[1] = src[1];
    pf[p] = u.v;
  }

  __syncthreads();

  v8f acc = {};
  for (int tb = 0; tb < T_; tb += PF) {
#pragma unroll
    for (int p = 0; p < PF; ++p) {
      const int t = tb + p;
      v8f cfrag = pf[p];
      if (t + PF < T_) {  // refill pipeline slot
        const v4f* src = (const v4f*)(xpbase + (size_t)(t + PF) * 4096);
        PUN8 u; u.q[0] = src[0]; u.q[1] = src[1];
        pf[p] = u.v;
      }
      if (t + 96 < T_)  // pull far-future tiles into L2
        __builtin_prefetch((const void*)(xpbase + (size_t)(t + 96) * 4096), 0, 1);

      // A fragments (current h) from LDS, already in 16-bit A layout:
      // lane holds row r, chunks {kb+koff+0..7} and {kb+16+koff+0..7}.
      const int cur = t & 1;
      const v8h* hrow = (const v8h*)(&hbuf[cur][r * 64]);
      union PUNA { v16h v; v8h h[2]; } a0, a1;
      a0.h[0] = hrow[(0 + koff) >> 3];
      a0.h[1] = hrow[(16 + koff) >> 3];
      a1.h[0] = hrow[(32 + koff) >> 3];
      a1.h[1] = hrow[(48 + koff) >> 3];

      // D = A1*B1 + (A0*B0 + xp)   (xp rides in for free as the C operand)
      acc = __builtin_amdgcn_wmma_f32_16x16x32_f16(
          false, a0.v, false, Bf[0], (short)0, cfrag, false, false);
      acc = __builtin_amdgcn_wmma_f32_16x16x32_f16(
          false, a1.v, false, Bf[1], (short)0, acc, false, false);

      // tanh, write h_new (f16) into the other buffer in row-major [16][64]
      _Float16* hout = &hbuf[cur ^ 1][0];
#pragma unroll
      for (int v = 0; v < 8; ++v) {
        float tv = fast_tanh(acc[v]);
        acc[v] = tv;
        hout[(v + koff) * 64 + nglob] = (_Float16)tv;
      }
      __syncthreads();
    }
  }

  // Final hidden state (f32, post-tanh) for the output head.
#pragma unroll
  for (int v = 0; v < 8; ++v)
    hfinal[(mblk * 16 + v + koff) * H_ + nglob] = acc[v];
}

// ---------------------------------------------------------------------------
// Kernel 3: out[b,c] = h_T[b,:] . W_out[c,:] + b_out[c]
// ---------------------------------------------------------------------------
__global__ __launch_bounds__(640) void out_kernel(
    const float* __restrict__ hfinal, const float* __restrict__ W_out,
    const float* __restrict__ b_out, float* __restrict__ out) {
  int tid = threadIdx.x;
  if (tid >= B_ * C_) return;
  int b = tid / C_, c = tid - b * C_;
  float s = b_out[c];
  const float* hr = hfinal + b * H_;
  const float* wr = W_out + c * H_;
#pragma unroll
  for (int h = 0; h < H_; ++h) s = fmaf(hr[h], wr[h], s);
  out[tid] = s;
}

extern "C" void kernel_launch(void* const* d_in, const int* in_sizes, int n_in,
                              void* d_out, int out_size, void* d_ws, size_t ws_size,
                              hipStream_t stream) {
  const float* input = (const float*)d_in[0];
  const float* W_ih  = (const float*)d_in[1];
  const float* W_hh  = (const float*)d_in[2];
  const float* b_ih  = (const float*)d_in[3];
  const float* b_hh  = (const float*)d_in[4];
  const float* W_out = (const float*)d_in[5];
  const float* b_out = (const float*)d_in[6];
  float* ws = (float*)d_ws;
  float* hfinal = ws + (size_t)T_ * 4096;

  xproj_kernel<<<T_, 256, 0, stream>>>(input, W_ih, b_ih, b_hh, ws);
  rnn_scan_kernel<<<4, 128, 0, stream>>>(W_hh, ws, hfinal);
  out_kernel<<<1, 640, 0, stream>>>(hfinal, W_out, b_out, (float*)d_out);
}
